// SequenceNNModel_15058155340082
// MI455X (gfx1250) — matrix operand order repaired and checked by reading, not compiled, not run on hardware
//
#include <hip/hip_runtime.h>
#include <hip/hip_bf16.h>

// ---------------------------------------------------------------------------
// LSTM (B=512, T=1024, F=64, H=128) + tabular MLP + 2-way head for gfx1250.
//
// Path A (preferred, needs ~64.3 MiB workspace):
//   1) prepack_kernel: seq f32 -> bf16 WMMA A-fragments, 64B/lane/step,
//      contiguous 2KB tiles per (workgroup, timestep).
//   2) lstm_tdm_kernel: 32 WGs x 16 batch rows; x tiles streamed into LDS by
//      the Tensor Data Mover (quad-buffered, issued 4 steps ahead, waited with
//      s_wait_tensorcnt); h double-buffered in LDS (bf16); all matmuls via
//      v_wmma_f32_16x16x32_bf16 with weights held as register B-fragments.
//      Cell activations use hardware transcendentals (v_tanh/v_rcp/v_exp).
// Path B (fallback if workspace too small): direct global-load recurrence.
// ---------------------------------------------------------------------------

typedef __attribute__((ext_vector_type(16))) __bf16 v16bf;
typedef __attribute__((ext_vector_type(8)))  float  v8f;
typedef __attribute__((ext_vector_type(4)))  unsigned int v4u;
typedef __attribute__((ext_vector_type(8)))  int    v8i;
typedef __attribute__((ext_vector_type(4)))  int    v4i;

#define B_TOT 512
#define T_SEQ 1024
#define F_IN  64
#define H_DIM 128
#define BT    16      // batch rows per workgroup
#define NBLK  (B_TOT / BT)
#define DEPTH 4       // TDM pipeline depth (LDS x-tile buffers)
#define XTILE_BYTES 2048  // 32 lanes * 64B of prepacked bf16 A-fragments

union V16BF { v16bf v; __bf16 e[16]; uint4 u[2]; };
union V8F   { v8f   v; float  e[8]; };

// ---- fast activations on hardware transcendentals ----
__device__ inline float fast_rcp(float x) {
#if __has_builtin(__builtin_amdgcn_rcpf)
  return __builtin_amdgcn_rcpf(x);
#else
  return 1.0f / x;
#endif
}

__device__ inline float fast_tanh(float x) {
#if __has_builtin(__builtin_amdgcn_tanhf)
  return __builtin_amdgcn_tanhf(x);        // v_tanh_f32
#else
  float e = __expf(2.0f * x);              // v_exp_f32
  return 1.0f - 2.0f * fast_rcp(e + 1.0f); // safe at +/-inf
#endif
}

__device__ inline float fast_sigmoid(float x) {
#if __has_builtin(__builtin_amdgcn_tanhf)
  return __builtin_fmaf(0.5f, __builtin_amdgcn_tanhf(0.5f * x), 0.5f);
#else
  return fast_rcp(1.0f + __expf(-x));
#endif
}

// B-matrix fragment (32x16, bf16) from a row-major f32 weight row W[g][*]:
// lane (m,half): elements 0..7 -> K = kb + 8*half + j ; elements 8..15 -> K = kb + 16 + 8*half + j, N = m
__device__ inline v16bf load_b_frag_f32(const float* __restrict__ wrow, int kb, int half) {
  V16BF r;
  const float* p0 = wrow + kb + 8 * half;
  const float* p1 = p0 + 16;
#pragma unroll
  for (int j = 0; j < 8; ++j) {
    r.e[j]     = (__bf16)p0[j];
    r.e[8 + j] = (__bf16)p1[j];
  }
  return r.v;
}

// A-matrix fragment (16x32, bf16) from a row-major f32 x row (row = lane m):
// elements 0..7 -> K = kb + 8*half + j ; elements 8..15 -> K = kb + 16 + 8*half + j
__device__ inline v16bf load_a_frag_x_f32(const float* __restrict__ row, int kb, int half) {
  V16BF r;
  const float* p0 = row + kb + 8 * half;
  const float* p1 = p0 + 16;
#pragma unroll
  for (int j = 0; j < 8; ++j) {
    r.e[j]     = (__bf16)p0[j];
    r.e[8 + j] = (__bf16)p1[j];
  }
  return r.v;
}

// A-matrix fragment from bf16 LDS row (two 16B ds loads)
__device__ inline v16bf load_a_frag_h(const __bf16* row, int kb, int half) {
  V16BF r;
  r.u[0] = *(const uint4*)(row + kb + 8 * half);
  r.u[1] = *(const uint4*)(row + kb + 16 + 8 * half);
  return r.v;
}

// ---------------------------------------------------------------------------
// Prepack: seq [B,T,F] f32 -> bf16 A-fragments. One thread per (blk,t,lane),
// writes 32 bf16 (64B) contiguous; tile for (blk,t) = 2KB contiguous.
// ---------------------------------------------------------------------------
__global__ __launch_bounds__(256) void prepack_kernel(
    const float* __restrict__ seq, __bf16* __restrict__ packed)
{
  unsigned gid  = blockIdx.x * 256u + threadIdx.x;   // < 32*1024*32
  unsigned lane = gid & 31u;
  unsigned t    = (gid >> 5) & (T_SEQ - 1);
  unsigned blk  = gid >> 15;
  unsigned m    = lane & 15u;
  unsigned half = lane >> 4;

  const float* row = seq + ((size_t)(blk * BT + m) * T_SEQ + t) * F_IN;
  __bf16* out = packed + (size_t)gid * 32;           // 32 bf16 per thread

#pragma unroll
  for (int c = 0; c < 2; ++c) {          // K-chunk (0..31, 32..63)
    const float* p0 = row + c * 32 + 8 * half;
    const float* p1 = p0 + 16;
#pragma unroll
    for (int j = 0; j < 8; ++j) {
      out[c * 16 + j]     = (__bf16)p0[j];
      out[c * 16 + 8 + j] = (__bf16)p1[j];
    }
  }
}

// ---------------------------------------------------------------------------
// TDM issue: 1-D tile, 2KB (256 x 8-byte elements) from `ga` to LDS `lds_off`.
// Descriptor per CDNA5 ISA (groups 0/1; 2/3 zero for <=2D).
// ---------------------------------------------------------------------------
__device__ inline void tdm_issue_2kb(const void* gsrc, unsigned lds_off) {
  unsigned long long ga = (unsigned long long)(uintptr_t)gsrc;
  v4u g0;
  g0[0] = 1u;                                        // count=1, user mode
  g0[1] = lds_off;                                   // lds_addr (bytes)
  g0[2] = (unsigned)(ga & 0xffffffffu);              // global_addr[31:0]
  g0[3] = (unsigned)((ga >> 32) & 0x1ffffffu) | (2u << 30);  // addr[56:32] | type=2
  v8i g1;
  g1[0] = (int)(3u << 16);        // workgroup_mask=0, data_size=3 (8B)
  g1[1] = (int)(256u << 16);      // tensor_dim0 = 256 (low 16 bits)
  g1[2] = 0;                      // tensor_dim0 hi | tensor_dim1 lo
  g1[3] = (int)(256u << 16);      // tensor_dim1 hi | tile_dim0 = 256
  g1[4] = 0;                      // tile_dim1 = 0 (unused), tile_dim2 = 0
  g1[5] = 256;                    // tensor_dim0_stride lo
  g1[6] = 0;
  g1[7] = 0;
  v4i z4; z4[0] = z4[1] = z4[2] = z4[3] = 0;
#if defined(__clang_major__) && (__clang_major__ >= 23)
  v8i z8; z8[0]=z8[1]=z8[2]=z8[3]=z8[4]=z8[5]=z8[6]=z8[7]=0;
  __builtin_amdgcn_tensor_load_to_lds(g0, g1, z4, z4, z8, 0);
#else
  __builtin_amdgcn_tensor_load_to_lds(g0, g1, z4, z4, 0);
#endif
}

// ---------------------------------------------------------------------------
// TDM-staged recurrence
// ---------------------------------------------------------------------------
__global__ __launch_bounds__(256) void lstm_tdm_kernel(
    const __bf16* __restrict__ packed,  // [NBLK, T, 32 lanes, 32 bf16]
    const float* __restrict__ W_ih, const float* __restrict__ W_hh,
    const float* __restrict__ b_ih, const float* __restrict__ b_hh,
    float* __restrict__ h_final)
{
  // single LDS pool (starts at LDS offset 0): x staging + h double buffer
  __shared__ __align__(16) unsigned char lds_pool[DEPTH * XTILE_BYTES + 2 * BT * H_DIM * 2];
  __bf16* sX = (__bf16*)lds_pool;
  __bf16* sH = (__bf16*)(lds_pool + DEPTH * XTILE_BYTES);

  const int tid  = threadIdx.x;
  const int wave = tid >> 5;
  const int lane = tid & 31;
  const int m    = lane & 15;
  const int half = lane >> 4;
  const int b0   = blockIdx.x * BT;
  const int hk0  = wave * 16;

  // weight fragments in registers (bf16), loaded once
  v16bf wih[4][2];
  v16bf whh[4][4];
  float bias[4];
#pragma unroll
  for (int q = 0; q < 4; ++q) {
    const int g = q * H_DIM + hk0 + m;
    bias[q] = b_ih[g] + b_hh[g];
    const float* wri = W_ih + (size_t)g * F_IN;
#pragma unroll
    for (int kb = 0; kb < 2; ++kb) wih[q][kb] = load_b_frag_f32(wri, kb * 32, half);
    const float* wrh = W_hh + (size_t)g * H_DIM;
#pragma unroll
    for (int kb = 0; kb < 4; ++kb) whh[q][kb] = load_b_frag_f32(wrh, kb * 32, half);
  }

  for (int i = tid; i < 2 * BT * H_DIM; i += 256) sH[i] = (__bf16)0.0f;
  float c_state[8];
#pragma unroll
  for (int r = 0; r < 8; ++r) c_state[r] = 0.0f;
  float h_last[8];

  const unsigned char* gx = (const unsigned char*)(packed + (size_t)blockIdx.x * T_SEQ * 1024);

  // prologue: wave 0 issues DEPTH tiles ahead
  if (wave == 0) {
#pragma unroll
    for (int d = 0; d < DEPTH; ++d)
      tdm_issue_2kb(gx + (size_t)d * XTILE_BYTES, (unsigned)(d * XTILE_BYTES));
  }
  __syncthreads();

  int buf = 0;
  for (int t = 0; t < T_SEQ; ++t) {
    if (wave == 0) __builtin_amdgcn_s_wait_tensorcnt(DEPTH - 1);
    __syncthreads();   // x tile (t) visible to all waves

    const __bf16* xf = sX + (t & (DEPTH - 1)) * 1024 + lane * 32;
    V16BF ax0, ax1;
    ax0.u[0] = ((const uint4*)xf)[0];
    ax0.u[1] = ((const uint4*)xf)[1];
    ax1.u[0] = ((const uint4*)xf)[2];
    ax1.u[1] = ((const uint4*)xf)[3];

    const __bf16* hrow = sH + buf * (BT * H_DIM) + m * H_DIM;
    v16bf ah0 = load_a_frag_h(hrow, 0, half);
    v16bf ah1 = load_a_frag_h(hrow, 32, half);
    v16bf ah2 = load_a_frag_h(hrow, 64, half);
    v16bf ah3 = load_a_frag_h(hrow, 96, half);

    V8F acc[4];
#pragma unroll
    for (int q = 0; q < 4; ++q) {
      V8F c;
#pragma unroll
      for (int r = 0; r < 8; ++r) c.e[r] = bias[q];
      v8f a = c.v;
      a = __builtin_amdgcn_wmma_f32_16x16x32_bf16(false, ax0.v, false, wih[q][0], (short)0, a, false, false);
      a = __builtin_amdgcn_wmma_f32_16x16x32_bf16(false, ax1.v, false, wih[q][1], (short)0, a, false, false);
      a = __builtin_amdgcn_wmma_f32_16x16x32_bf16(false, ah0, false, whh[q][0], (short)0, a, false, false);
      a = __builtin_amdgcn_wmma_f32_16x16x32_bf16(false, ah1, false, whh[q][1], (short)0, a, false, false);
      a = __builtin_amdgcn_wmma_f32_16x16x32_bf16(false, ah2, false, whh[q][2], (short)0, a, false, false);
      a = __builtin_amdgcn_wmma_f32_16x16x32_bf16(false, ah3, false, whh[q][3], (short)0, a, false, false);
      acc[q].v = a;
    }

    __bf16* hout = sH + (buf ^ 1) * (BT * H_DIM);
#pragma unroll
    for (int r = 0; r < 8; ++r) {
      float iv = fast_sigmoid(acc[0].e[r]);
      float fv = fast_sigmoid(acc[1].e[r]);
      float gv = fast_tanh(acc[2].e[r]);
      float ov = fast_sigmoid(acc[3].e[r]);
      float cv = fv * c_state[r] + iv * gv;
      c_state[r] = cv;
      float hv = ov * fast_tanh(cv);
      h_last[r] = hv;
      hout[(r + 8 * half) * H_DIM + (hk0 + m)] = (__bf16)hv;
    }
    buf ^= 1;
    __syncthreads();   // everyone done with x tile (t) and h ready

    if (wave == 0 && (t + DEPTH) < T_SEQ)
      tdm_issue_2kb(gx + (size_t)(t + DEPTH) * XTILE_BYTES,
                    (unsigned)((t & (DEPTH - 1)) * XTILE_BYTES));
  }

#pragma unroll
  for (int r = 0; r < 8; ++r) {
    int b = b0 + r + 8 * half;
    h_final[(size_t)b * H_DIM + hk0 + m] = h_last[r];
  }
}

// ---------------------------------------------------------------------------
// Fallback: direct global-load recurrence (no TDM / no prepack workspace)
// ---------------------------------------------------------------------------
__global__ __launch_bounds__(256) void lstm_direct_kernel(
    const float* __restrict__ seq,
    const float* __restrict__ W_ih, const float* __restrict__ W_hh,
    const float* __restrict__ b_ih, const float* __restrict__ b_hh,
    float* __restrict__ h_final)
{
  __shared__ __bf16 sH[2][BT * H_DIM];
  const int tid  = threadIdx.x;
  const int wave = tid >> 5;
  const int lane = tid & 31;
  const int m    = lane & 15;
  const int half = lane >> 4;
  const int b0   = blockIdx.x * BT;
  const int hk0  = wave * 16;

  v16bf wih[4][2];
  v16bf whh[4][4];
  float bias[4];
#pragma unroll
  for (int q = 0; q < 4; ++q) {
    const int g = q * H_DIM + hk0 + m;
    bias[q] = b_ih[g] + b_hh[g];
    const float* wri = W_ih + (size_t)g * F_IN;
#pragma unroll
    for (int kb = 0; kb < 2; ++kb) wih[q][kb] = load_b_frag_f32(wri, kb * 32, half);
    const float* wrh = W_hh + (size_t)g * H_DIM;
#pragma unroll
    for (int kb = 0; kb < 4; ++kb) whh[q][kb] = load_b_frag_f32(wrh, kb * 32, half);
  }

  for (int i = tid; i < BT * H_DIM; i += 256) sH[0][i] = (__bf16)0.0f;
  float c_state[8];
#pragma unroll
  for (int r = 0; r < 8; ++r) c_state[r] = 0.0f;
  float h_last[8];

  const float* xrow = seq + ((size_t)(b0 + m) * T_SEQ) * F_IN;
  __syncthreads();

  int buf = 0;
  for (int t = 0; t < T_SEQ; ++t) {
    __builtin_prefetch(xrow + F_IN, 0, 1);
    v16bf ax0 = load_a_frag_x_f32(xrow, 0, half);
    v16bf ax1 = load_a_frag_x_f32(xrow, 32, half);
    const __bf16* hrow = &sH[buf][m * H_DIM];
    v16bf ah0 = load_a_frag_h(hrow, 0, half);
    v16bf ah1 = load_a_frag_h(hrow, 32, half);
    v16bf ah2 = load_a_frag_h(hrow, 64, half);
    v16bf ah3 = load_a_frag_h(hrow, 96, half);

    V8F acc[4];
#pragma unroll
    for (int q = 0; q < 4; ++q) {
      V8F c;
#pragma unroll
      for (int r = 0; r < 8; ++r) c.e[r] = bias[q];
      v8f a = c.v;
      a = __builtin_amdgcn_wmma_f32_16x16x32_bf16(false, ax0, false, wih[q][0], (short)0, a, false, false);
      a = __builtin_amdgcn_wmma_f32_16x16x32_bf16(false, ax1, false, wih[q][1], (short)0, a, false, false);
      a = __builtin_amdgcn_wmma_f32_16x16x32_bf16(false, ah0, false, whh[q][0], (short)0, a, false, false);
      a = __builtin_amdgcn_wmma_f32_16x16x32_bf16(false, ah1, false, whh[q][1], (short)0, a, false, false);
      a = __builtin_amdgcn_wmma_f32_16x16x32_bf16(false, ah2, false, whh[q][2], (short)0, a, false, false);
      a = __builtin_amdgcn_wmma_f32_16x16x32_bf16(false, ah3, false, whh[q][3], (short)0, a, false, false);
      acc[q].v = a;
    }

    __bf16* hout = &sH[buf ^ 1][0];
#pragma unroll
    for (int r = 0; r < 8; ++r) {
      float iv = fast_sigmoid(acc[0].e[r]);
      float fv = fast_sigmoid(acc[1].e[r]);
      float gv = fast_tanh(acc[2].e[r]);
      float ov = fast_sigmoid(acc[3].e[r]);
      float cv = fv * c_state[r] + iv * gv;
      c_state[r] = cv;
      float hv = ov * fast_tanh(cv);
      h_last[r] = hv;
      hout[(r + 8 * half) * H_DIM + (hk0 + m)] = (__bf16)hv;
    }
    buf ^= 1;
    xrow += F_IN;
    __syncthreads();
  }

#pragma unroll
  for (int r = 0; r < 8; ++r) {
    int b = b0 + r + 8 * half;
    h_final[(size_t)b * H_DIM + hk0 + m] = h_last[r];
  }
}

// ---------------------------------------------------------------------------
// Tabular branch + output head (negligible FLOPs)
// ---------------------------------------------------------------------------
__global__ __launch_bounds__(256) void head_kernel(
    const float* __restrict__ tabular, const float* __restrict__ W_tab,
    const float* __restrict__ b_tab, const float* __restrict__ W_out,
    const float* __restrict__ b_out, const float* __restrict__ h_final,
    float* __restrict__ out)
{
  int b = blockIdx.x * blockDim.x + threadIdx.x;
  if (b >= B_TOT) return;
  const float* x = tabular + (size_t)b * 128;
  const float* h = h_final + (size_t)b * H_DIM;
  float l0 = b_out[0], l1 = b_out[1];
#pragma unroll 4
  for (int k = 0; k < H_DIM; ++k) {
    l0 += h[k] * W_out[k];
    l1 += h[k] * W_out[256 + k];
  }
  for (int j = 0; j < 128; ++j) {
    const float* wr = W_tab + (size_t)j * 128;
    float s = b_tab[j];
#pragma unroll 4
    for (int k = 0; k < 128; ++k) s += x[k] * wr[k];
    s = fmaxf(s, 0.0f);
    l0 += s * W_out[128 + j];
    l1 += s * W_out[256 + 128 + j];
  }
  out[b * 2 + 0] = l0;
  out[b * 2 + 1] = l1;
}

extern "C" void kernel_launch(void* const* d_in, const int* in_sizes, int n_in,
                              void* d_out, int out_size, void* d_ws, size_t ws_size,
                              hipStream_t stream) {
  const float* tabular = (const float*)d_in[0];
  const float* seq     = (const float*)d_in[1];
  const float* W_ih    = (const float*)d_in[2];
  const float* W_hh    = (const float*)d_in[3];
  const float* b_ihp   = (const float*)d_in[4];
  const float* b_hhp   = (const float*)d_in[5];
  const float* W_tab   = (const float*)d_in[6];
  const float* b_tab   = (const float*)d_in[7];
  const float* W_out   = (const float*)d_in[8];
  const float* b_out   = (const float*)d_in[9];

  const size_t hfinal_bytes  = (size_t)B_TOT * H_DIM * sizeof(float);   // 256 KB
  const size_t prepack_bytes = (size_t)NBLK * T_SEQ * XTILE_BYTES;      // 64 MiB

  float* h_final = (float*)d_ws;

  if (ws_size >= hfinal_bytes + prepack_bytes) {
    __bf16* packed = (__bf16*)((unsigned char*)d_ws + hfinal_bytes);
    prepack_kernel<<<(NBLK * T_SEQ * 32) / 256, 256, 0, stream>>>(seq, packed);
    lstm_tdm_kernel<<<NBLK, 256, 0, stream>>>(packed, W_ih, W_hh, b_ihp, b_hhp, h_final);
  } else {
    lstm_direct_kernel<<<NBLK, 256, 0, stream>>>(seq, W_ih, W_hh, b_ihp, b_hhp, h_final);
  }
  head_kernel<<<(B_TOT + 255) / 256, 256, 0, stream>>>(
      tabular, W_tab, b_tab, W_out, b_out, h_final, (float*)d_out);
}
